// CausalSelfAttention_1322849927549
// MI455X (gfx1250) — compile-verified, matrix operand order checked by reading, not run on hardware
//
#include <hip/hip_runtime.h>
#include <hip/hip_bf16.h>

typedef __bf16 bf16_t;
typedef __bf16 bf16x16 __attribute__((ext_vector_type(16)));
typedef float  f32x8  __attribute__((ext_vector_type(8)));
typedef unsigned int u32x4 __attribute__((ext_vector_type(4)));
typedef int i32x4v __attribute__((ext_vector_type(4)));
typedef int i32x8v __attribute__((ext_vector_type(8)));

#define DIM    2048
#define NHEADS 16
#define HD     128
#define BATCH  2
#define SEQ    2048
#define ROWS   (BATCH * SEQ)   // 4096

// ---------------------------------------------------------------------------
// WMMA helpers
// ---------------------------------------------------------------------------
__device__ __forceinline__ f32x8 wmma_bf16(bf16x16 a, bf16x16 b, f32x8 c) {
  return __builtin_amdgcn_wmma_f32_16x16x32_bf16(
      /*neg_a=*/false, a, /*neg_b=*/false, b,
      /*c_mod=*/(short)0, c, /*reuse_a=*/false, /*reuse_b=*/false);
}

__device__ __forceinline__ f32x8 zero_f32x8() {
  f32x8 z;
#pragma unroll
  for (int e = 0; e < 8; ++e) z[e] = 0.0f;
  return z;
}

// A-layout 16x32 bf16 fragment from K-contiguous storage (elem (m,k) at
// base[m*stride+k]).  Also a B-matrix stored N-major (B(k,n) at base[n*stride+k]).
// ISA: lanes 0-15 m=lane, K in {0..7,16..23}; lanes 16-31 K in {8..15,24..31}.
__device__ __forceinline__ bf16x16 load_frag_k_contig(const bf16_t* base, int stride, int lane) {
  const int m = lane & 15;
  const int hl = (lane >> 4) & 1;
  const bf16_t* p = base + (size_t)m * stride + hl * 8;
  bf16x16 f;
#pragma unroll
  for (int e = 0; e < 16; ++e) {
    const int k = ((e & 8) << 1) + (e & 7);
    f[e] = p[k];
  }
  return f;
}

// B-layout 32x16 fragment from K-major storage (B(k,n) at base[k*stride+n]).
__device__ __forceinline__ bf16x16 load_frag_n_contig(const bf16_t* base, int stride, int lane) {
  const int n = lane & 15;
  const int hl = (lane >> 4) & 1;
  const bf16_t* p = base + n + (size_t)(hl * 8) * stride;
  bf16x16 f;
#pragma unroll
  for (int e = 0; e < 16; ++e) {
    const int k = ((e & 8) << 1) + (e & 7);
    f[e] = p[(size_t)k * stride];
  }
  return f;
}

// ---------------------------------------------------------------------------
// CDNA5 async copy: GLOBAL_LOAD_ASYNC_TO_LDS_B128 (GVS addressing).
// ldsOff: byte offset into workgroup LDS; vOff: 32-bit byte offset; base: SGPR64.
// ---------------------------------------------------------------------------
__device__ __forceinline__ void async_copy_b128(unsigned ldsOff, unsigned vOff,
                                                unsigned long long base) {
  asm volatile("global_load_async_to_lds_b128 %0, %1, %2"
               :: "v"(ldsOff), "v"(vOff), "s"(base) : "memory");
}

// ---------------------------------------------------------------------------
// CDNA5 Tensor Data Mover: 2D bf16 tile (rows x rowElems) -> LDS, with LDS
// padding of 4 DWORDs after every 64 DWORDs (256B row) => 136-elem LDS stride.
// ---------------------------------------------------------------------------
__device__ __forceinline__ void tdm_load_2d_bf16(unsigned ldsOff, const bf16_t* gptr,
                                                 int rows, int rowElems, int strideElems) {
  const unsigned long long ga = (unsigned long long)(uintptr_t)gptr;
  u32x4 g0;
  g0[0] = 1u;                                   // count=1 (valid user descriptor)
  g0[1] = ldsOff;                               // lds_addr
  g0[2] = (unsigned)ga;                         // global_addr[31:0]
  g0[3] = (unsigned)(ga >> 32) | (2u << 30);    // global_addr[56:32] | type=2
  i32x8v g1;
  g1[0] = (int)((1u << 16) |                    // data_size = 2 bytes
                (1u << 20) |                    // pad_enable
                (5u << 22) |                    // pad_interval: 64 DWORDs
                (3u << 25));                    // pad_amount: 4 DWORDs
  g1[1] = (int)(((unsigned)rowElems) << 16);    // tensor_dim0[15:0] (hi half of dw)
  g1[2] = (int)(((unsigned)SEQ) << 16);         // tensor_dim0[31:16]=0 | tensor_dim1[15:0]
  g1[3] = (int)(((unsigned)rowElems) << 16);    // tensor_dim1[31:16]=0 | tile_dim0
  g1[4] = rows;                                 // tile_dim1 | tile_dim2=0
  g1[5] = strideElems;                          // tensor_dim0_stride[31:0]
  g1[6] = 0;                                    // stride hi | tensor_dim1_stride lo
  g1[7] = 0;
  i32x4v gz;
  gz[0] = 0; gz[1] = 0; gz[2] = 0; gz[3] = 0;
#if defined(__clang_major__) && (__clang_major__ >= 23)
  i32x8v gz8;
#pragma unroll
  for (int e = 0; e < 8; ++e) gz8[e] = 0;
  __builtin_amdgcn_tensor_load_to_lds(g0, g1, gz, gz, gz8, 0);
#else
  __builtin_amdgcn_tensor_load_to_lds(g0, g1, gz, gz, 0);
#endif
}

// ---------------------------------------------------------------------------
// f32 -> bf16 conversion
// ---------------------------------------------------------------------------
__global__ void cvt_bf16_kernel(const float* __restrict__ in, bf16_t* __restrict__ out, int n) {
  const int i = blockIdx.x * blockDim.x + threadIdx.x;
  if (i < n) out[i] = (bf16_t)in[i];
}

// ---------------------------------------------------------------------------
// Tiled WMMA GEMM:  C[M x N] = A[M x K] * W[N x K]^T
// 128x128 block tile, K-step 32, 256 threads (8 waves, 4x2 wave grid of 32x64).
// Double-buffered LDS filled with async-to-LDS copies overlapped with WMMA.
// mode 0: scatter into per-head q/k/v bf16 buffers ([B,H,T,hd]); N = 3*DIM.
// mode 1: write f32 to outf (row-major, ldc = N).
// ---------------------------------------------------------------------------
__device__ __forceinline__ void gemm_stage_async(const bf16_t* A, const bf16_t* W,
                                                 int Kd, int bm, int bn, int k0, int tid,
                                                 bf16_t* aBuf, bf16_t* wBuf) {
  const unsigned long long aBase = (unsigned long long)(uintptr_t)A;
  const unsigned long long wBase = (unsigned long long)(uintptr_t)W;
#pragma unroll
  for (int it = 0; it < 2; ++it) {
    const int i = tid + it * 256;               // 512 chunks of 8 bf16 per tile
    const int row = i >> 2, c8 = (i & 3) << 3;
    const unsigned aL = (unsigned)(uintptr_t)&aBuf[row * 40 + c8];
    const unsigned wL = (unsigned)(uintptr_t)&wBuf[row * 40 + c8];
    const unsigned aG = (unsigned)(((unsigned)(bm + row) * (unsigned)Kd + k0 + c8) * 2u);
    const unsigned wG = (unsigned)(((unsigned)(bn + row) * (unsigned)Kd + k0 + c8) * 2u);
    async_copy_b128(aL, aG, aBase);
    async_copy_b128(wL, wG, wBase);
  }
}

__global__ __launch_bounds__(256) void gemm_bf16_kernel(
    const bf16_t* __restrict__ A, const bf16_t* __restrict__ W,
    int Kd, int N, int mode,
    bf16_t* __restrict__ qb, bf16_t* __restrict__ kb, bf16_t* __restrict__ vb,
    float* __restrict__ outf) {
  __shared__ bf16_t aL[2][128 * 40];   // 32 K + 8 pad, double buffered
  __shared__ bf16_t wL[2][128 * 40];

  const int tid  = threadIdx.x;
  const int wave = tid >> 5, lane = tid & 31;
  const int wr = wave >> 1, wc = wave & 1;
  const int bm = blockIdx.y * 128, bn = blockIdx.x * 128;

  f32x8 acc[2][4];
#pragma unroll
  for (int i = 0; i < 2; ++i)
#pragma unroll
    for (int j = 0; j < 4; ++j) acc[i][j] = zero_f32x8();

  const int nsteps = Kd >> 5;
  gemm_stage_async(A, W, Kd, bm, bn, 0, tid, aL[0], wL[0]);

  for (int st = 0; st < nsteps; ++st) {
    if (st + 1 < nsteps) {
      gemm_stage_async(A, W, Kd, bm, bn, (st + 1) << 5, tid,
                       aL[(st + 1) & 1], wL[(st + 1) & 1]);
      // async loads complete in order: cnt<=4 => the 4 ops of stage `st` landed
      asm volatile("s_wait_asynccnt 0x4" ::: "memory");
    } else {
      asm volatile("s_wait_asynccnt 0x0" ::: "memory");
    }
    __syncthreads();

    const bf16_t* aC = aL[st & 1];
    const bf16_t* wC = wL[st & 1];
    bf16x16 af[2], bfr[4];
#pragma unroll
    for (int i = 0; i < 2; ++i)
      af[i] = load_frag_k_contig(&aC[(wr * 32 + i * 16) * 40], 40, lane);
#pragma unroll
    for (int j = 0; j < 4; ++j)
      bfr[j] = load_frag_k_contig(&wC[(wc * 64 + j * 16) * 40], 40, lane);
#pragma unroll
    for (int i = 0; i < 2; ++i)
#pragma unroll
      for (int j = 0; j < 4; ++j) acc[i][j] = wmma_bf16(af[i], bfr[j], acc[i][j]);
    __syncthreads();
  }

  const int n_ = lane & 15, hl = (lane >> 4) & 1;
#pragma unroll
  for (int i = 0; i < 2; ++i)
#pragma unroll
    for (int j = 0; j < 4; ++j)
#pragma unroll
      for (int r = 0; r < 8; ++r) {
        const int row = bm + wr * 32 + i * 16 + r + 8 * hl;
        const int col = bn + wc * 64 + j * 16 + n_;
        const float val = acc[i][j][r];
        if (mode == 0) {
          const int sec = col >> 11, rem = col & (DIM - 1);
          const int head = rem >> 7, d = rem & (HD - 1);
          const int b = row >> 11, t = row & (SEQ - 1);
          const size_t idx = (((size_t)(b * NHEADS + head) * SEQ) + t) * HD + d;
          bf16_t* dst = (sec == 0) ? qb : (sec == 1) ? kb : vb;
          dst[idx] = (bf16_t)val;
        } else {
          outf[(size_t)row * N + col] = val;
        }
      }
}

// ---------------------------------------------------------------------------
// RoPE (in place, q and k): one thread per (b,h,t,d2<64) pair
// ---------------------------------------------------------------------------
__global__ void rope_kernel(bf16_t* __restrict__ Q, bf16_t* __restrict__ K) {
  const int idx = blockIdx.x * blockDim.x + threadIdx.x;  // [0, B*H*T*64)
  const int d2 = idx & 63;
  const int t  = (idx >> 6) & (SEQ - 1);
  const int bh = idx >> 17;
  const size_t base = ((size_t)bh * SEQ + t) * HD;
  const float inv = __powf(10000.0f, -(float)(2 * d2) * (1.0f / (float)HD));
  const float fr = (float)t * inv;
  const float c = __cosf(fr), s = __sinf(fr);
  {
    const float a = (float)Q[base + d2], b = (float)Q[base + d2 + 64];
    Q[base + d2]      = (bf16_t)(a * c - b * s);
    Q[base + d2 + 64] = (bf16_t)(b * c + a * s);
  }
  {
    const float a = (float)K[base + d2], b = (float)K[base + d2 + 64];
    K[base + d2]      = (bf16_t)(a * c - b * s);
    K[base + d2 + 64] = (bf16_t)(b * c + a * s);
  }
}

// ---------------------------------------------------------------------------
// Causal flash attention. Grid: (SEQ/64, B*H); 128 threads = 4 waves.
// K/V 64x128 bf16 tiles DMA'd into LDS by the Tensor Data Mover (wave 0),
// completion via s_wait_tensorcnt + workgroup barrier. Each wave owns a
// 16-row Q tile. Output written as [B, T, H*hd] bf16.
// ---------------------------------------------------------------------------
__global__ __launch_bounds__(128) void flash_kernel(
    const bf16_t* __restrict__ Q, const bf16_t* __restrict__ K,
    const bf16_t* __restrict__ V, bf16_t* __restrict__ O) {
  __shared__ bf16_t kLds[64 * 136];      // [krow][d], TDM pad gives stride 136
  __shared__ bf16_t vLds[64 * 136];
  __shared__ bf16_t pLds[4][16 * 72];    // per-wave P staging, pad 8

  const int tid  = threadIdx.x;
  const int wave = tid >> 5, lane = tid & 31;
  const int q0 = blockIdx.x * 64;
  const int bh = blockIdx.y;
  const int b = bh >> 4, h = bh & 15;
  const int nlo = lane & 15, hl = (lane >> 4) & 1;

  const bf16_t* qBase = Q + ((size_t)bh * SEQ + q0 + wave * 16) * HD;
  bf16x16 qf[4];
#pragma unroll
  for (int kc = 0; kc < 4; ++kc) qf[kc] = load_frag_k_contig(qBase + kc * 32, HD, lane);

  f32x8 o[8];
  float mi[8], li[8];
#pragma unroll
  for (int dt = 0; dt < 8; ++dt) o[dt] = zero_f32x8();
#pragma unroll
  for (int r = 0; r < 8; ++r) { mi[r] = -3.0e38f; li[r] = 0.0f; }

  const int wave_qmax = q0 + wave * 16 + 15;
  const float scale = 0.08838834764831843f;  // 1/sqrt(128)

  const unsigned kLdsOff = (unsigned)(uintptr_t)&kLds[0];
  const unsigned vLdsOff = (unsigned)(uintptr_t)&vLds[0];

  for (int k0 = 0; k0 <= q0 + 63; k0 += 64) {
    if (wave == 0) {
      tdm_load_2d_bf16(kLdsOff, K + ((size_t)bh * SEQ + k0) * HD, 64, HD, HD);
      tdm_load_2d_bf16(vLdsOff, V + ((size_t)bh * SEQ + k0) * HD, 64, HD, HD);
      __builtin_amdgcn_s_wait_tensorcnt(0);
    }
    __syncthreads();

    if (k0 <= wave_qmax) {  // fully masked tiles skipped (barriers stay uniform)
      f32x8 s[4];
#pragma unroll
      for (int nt = 0; nt < 4; ++nt) {
        s[nt] = zero_f32x8();
#pragma unroll
        for (int kc = 0; kc < 4; ++kc)
          s[nt] = wmma_bf16(qf[kc],
                            load_frag_k_contig(&kLds[(nt * 16) * 136 + kc * 32], 136, lane),
                            s[nt]);
      }

#pragma unroll
      for (int nt = 0; nt < 4; ++nt)
#pragma unroll
        for (int r = 0; r < 8; ++r) {
          const int kcol = k0 + nt * 16 + nlo;
          const int qrow = q0 + wave * 16 + r + 8 * hl;
          float vv = s[nt][r] * scale;
          if (kcol > qrow) vv = -3.0e38f;
          s[nt][r] = vv;
        }

#pragma unroll
      for (int r = 0; r < 8; ++r) {
        float mt = fmaxf(fmaxf(s[0][r], s[1][r]), fmaxf(s[2][r], s[3][r]));
#pragma unroll
        for (int off = 8; off >= 1; off >>= 1) mt = fmaxf(mt, __shfl_xor(mt, off, 32));
        const float mnew = fmaxf(mi[r], mt);
        const float alpha = __expf(mi[r] - mnew);
        mi[r] = mnew;
        float rs = 0.0f;
#pragma unroll
        for (int nt = 0; nt < 4; ++nt) {
          const float p = __expf(s[nt][r] - mnew);
          s[nt][r] = p;
          rs += p;
        }
#pragma unroll
        for (int off = 8; off >= 1; off >>= 1) rs += __shfl_xor(rs, off, 32);
        li[r] = li[r] * alpha + rs;
#pragma unroll
        for (int dt = 0; dt < 8; ++dt) o[dt][r] *= alpha;
      }

      // stage P (C layout -> row-major bf16) in per-wave LDS, reload as A frags
      bf16_t* pw = &pLds[wave][0];
#pragma unroll
      for (int nt = 0; nt < 4; ++nt)
#pragma unroll
        for (int r = 0; r < 8; ++r)
          pw[(r + 8 * hl) * 72 + nt * 16 + nlo] = (bf16_t)s[nt][r];

      bf16x16 pf[2];
#pragma unroll
      for (int nc = 0; nc < 2; ++nc) pf[nc] = load_frag_k_contig(pw + nc * 32, 72, lane);

#pragma unroll
      for (int dt = 0; dt < 8; ++dt)
#pragma unroll
        for (int nc = 0; nc < 2; ++nc)
          o[dt] = wmma_bf16(pf[nc],
                            load_frag_n_contig(&vLds[(nc * 32) * 136 + dt * 16], 136, lane),
                            o[dt]);
    }
    __syncthreads();
  }

#pragma unroll
  for (int dt = 0; dt < 8; ++dt)
#pragma unroll
    for (int r = 0; r < 8; ++r) {
      const int t = q0 + wave * 16 + r + 8 * hl;
      O[((size_t)(b * SEQ + t)) * DIM + h * HD + dt * 16 + nlo] =
          (bf16_t)(o[dt][r] * (1.0f / li[r]));
    }
}

// ---------------------------------------------------------------------------
// Host launcher
// ---------------------------------------------------------------------------
extern "C" void kernel_launch(void* const* d_in, const int* in_sizes, int n_in,
                              void* d_out, int out_size, void* d_ws, size_t ws_size,
                              hipStream_t stream) {
  (void)in_sizes; (void)n_in; (void)out_size; (void)ws_size;

  const float* x     = (const float*)d_in[0];
  const float* w_qkv = (const float*)d_in[1];
  const float* w_out = (const float*)d_in[2];
  float* out = (float*)d_out;

  char* ws = (char*)d_ws;
  size_t off = 0;
  auto alloc = [&](size_t bytes) -> char* {
    char* p = ws + off;
    off += (bytes + 255) & ~(size_t)255;
    return p;
  };
  bf16_t* xb    = (bf16_t*)alloc((size_t)ROWS * DIM * 2);
  bf16_t* wqkvb = (bf16_t*)alloc((size_t)3 * DIM * DIM * 2);
  bf16_t* woutb = (bf16_t*)alloc((size_t)DIM * DIM * 2);
  bf16_t* qb    = (bf16_t*)alloc((size_t)ROWS * DIM * 2);
  bf16_t* kb    = (bf16_t*)alloc((size_t)ROWS * DIM * 2);
  bf16_t* vb    = (bf16_t*)alloc((size_t)ROWS * DIM * 2);
  bf16_t* attnb = (bf16_t*)alloc((size_t)ROWS * DIM * 2);

  const int nx    = ROWS * DIM;
  const int nwqkv = 3 * DIM * DIM;
  const int nwout = DIM * DIM;
  cvt_bf16_kernel<<<(nx    + 255) / 256, 256, 0, stream>>>(x,     xb,    nx);
  cvt_bf16_kernel<<<(nwqkv + 255) / 256, 256, 0, stream>>>(w_qkv, wqkvb, nwqkv);
  cvt_bf16_kernel<<<(nwout + 255) / 256, 256, 0, stream>>>(w_out, woutb, nwout);

  gemm_bf16_kernel<<<dim3(3 * DIM / 128, ROWS / 128), 256, 0, stream>>>(
      xb, wqkvb, DIM, 3 * DIM, /*mode=*/0, qb, kb, vb, nullptr);

  rope_kernel<<<(BATCH * NHEADS * SEQ * 64) / 256, 256, 0, stream>>>(qb, kb);

  flash_kernel<<<dim3(SEQ / 64, BATCH * NHEADS), 128, 0, stream>>>(qb, kb, vb, attnb);

  gemm_bf16_kernel<<<dim3(DIM / 128, ROWS / 128), 256, 0, stream>>>(
      attnb, woutb, DIM, DIM, /*mode=*/1, nullptr, nullptr, nullptr, out);
}
